// LSTMClassifier_13572096656118
// MI455X (gfx1250) — compile-verified, matrix operand order checked by reading, not run on hardware
//
#include <hip/hip_runtime.h>
#include <hip/hip_bf16.h>
#include <stdint.h>

// Problem dimensions
#define EMB_D   256
#define HID_D   512
#define G4_D    2048          // 4*H
#define BATCH_D 64
#define SEQ_D   512
#define OUT_D   18
#define MTOT    (BATCH_D * SEQ_D)   // 32768
#define NWG_LSTM 16

typedef __attribute__((ext_vector_type(16))) __bf16       v16bf;
typedef __attribute__((ext_vector_type(8)))  float        v8f;
typedef __attribute__((ext_vector_type(4)))  unsigned int u32x4;
typedef __attribute__((ext_vector_type(8)))  int          i32x8;
typedef __attribute__((ext_vector_type(4)))  int          i32x4;
typedef __attribute__((ext_vector_type(4)))  float        f32x4;

struct U256 { u32x4 lo; u32x4 hi; };

// ---------- helpers ----------
__device__ inline unsigned short f2bf(float x) {
    unsigned u = __builtin_bit_cast(unsigned, x);
    unsigned r = (u + 0x7FFFu + ((u >> 16) & 1u)) >> 16;   // RNE
    return (unsigned short)r;
}

__device__ inline float sigm_f(float x) { return 1.0f / (1.0f + __expf(-x)); }
__device__ inline float tanh_f(float x) { float e = __expf(2.0f * x); return 1.0f - 2.0f / (e + 1.0f); }

// Load a 16x32 bf16 WMMA A/B fragment: this lane's 8-elem chunk at `base`
// (K = kb..kb+7) and the second chunk at +16 elements (K = kb+16..kb+23).
__device__ inline v16bf ld_frag_g(const unsigned short* base) {
    U256 t;
    t.lo = *(const u32x4*)(base);
    t.hi = *(const u32x4*)(base + 16);
    return __builtin_bit_cast(v16bf, t);
}

__device__ inline v8f wmma_bf16(v16bf a, v16bf b, v8f c) {
    return __builtin_amdgcn_wmma_f32_16x16x32_bf16(false, a, false, b, (short)0, c, false, false);
}

// ---------- prep: f32 -> bf16 weight conversion ----------
__global__ void k_cvt_bf16(const float* __restrict__ src, unsigned short* __restrict__ dst, int n) {
    int i = blockIdx.x * blockDim.x + threadIdx.x;
    if (i < n) dst[i] = f2bf(src[i]);
}

// ---------- prep: embedding gather + bf16 convert ----------
// one thread per (row m, chunk of 8 cols): MTOT*32 threads
__global__ void k_gather(const int* __restrict__ x, const float* __restrict__ emb,
                         unsigned short* __restrict__ dst) {
    int tid = blockIdx.x * blockDim.x + threadIdx.x;
    int m = tid >> 5;
    int k = (tid & 31) << 3;
    int tok = x[m];
    const float* s = emb + (size_t)tok * EMB_D + k;
    unsigned short t[8];
#pragma unroll
    for (int j = 0; j < 8; j++) t[j] = f2bf(s[j]);
    u32x4 o;
    o[0] = (unsigned)t[0] | ((unsigned)t[1] << 16);
    o[1] = (unsigned)t[2] | ((unsigned)t[3] << 16);
    o[2] = (unsigned)t[4] | ((unsigned)t[5] << 16);
    o[3] = (unsigned)t[6] | ((unsigned)t[7] << 16);
    *(u32x4*)(dst + (size_t)m * EMB_D + k) = o;
}

// ---------- input projection: xproj[m,n] = e_bf16[m,:]·W_ih[n,:] + b_ih[n] + b_hh[n] ----------
// block = 256 thr (8 waves) -> 128x128 tile. wave w owns rows w*16..+16, all 128 N.
__global__ void __launch_bounds__(256) k_inproj(const unsigned short* __restrict__ eb,
                                                const unsigned short* __restrict__ wih,
                                                const float* __restrict__ b_ih,
                                                const float* __restrict__ b_hh,
                                                float* __restrict__ xproj) {
    int w = threadIdx.x >> 5;
    int lane = threadIdx.x & 31;
    int l16 = lane & 15;
    int hi = lane >> 4;
    int mbase = blockIdx.y * 128 + w * 16;
    int nbase = blockIdx.x * 128;

    v8f acc[8] = {};
    for (int kk = 0; kk < EMB_D / 32; kk++) {
        int kb = kk * 32 + hi * 8;
        v16bf a = ld_frag_g(eb + (size_t)(mbase + l16) * EMB_D + kb);
#pragma unroll
        for (int nt = 0; nt < 8; nt++) {
            v16bf b = ld_frag_g(wih + (size_t)(nbase + nt * 16 + l16) * EMB_D + kb);
            acc[nt] = wmma_bf16(a, b, acc[nt]);
        }
    }
#pragma unroll
    for (int nt = 0; nt < 8; nt++) {
        int n = nbase + nt * 16 + l16;
        float bias = b_ih[n] + b_hh[n];
#pragma unroll
        for (int r = 0; r < 8; r++) {
            int m = mbase + r + hi * 8;
            xproj[(size_t)m * G4_D + n] = acc[nt][r] + bias;
        }
    }
}

// ---------- persistent LSTM recurrence ----------
// 16 WGs x 256 thr. WG `wg` owns hidden units [wg*32, wg*32+32): computes gate
// columns {g*512 + wg*32 + nn} for g=0..3. W_hh slice (128 rows x 512 K, bf16)
// lives in LDS, loaded once via the Tensor Data Mover.
__global__ void __launch_bounds__(256) k_lstm(const unsigned short* __restrict__ whh,
                                              const float* __restrict__ xproj,
                                              unsigned short* __restrict__ hbf,  // 2 ping-pong buffers [64][512]
                                              float* __restrict__ hseq,          // [64][512][512]
                                              unsigned int* __restrict__ counter) {
    __shared__ unsigned short Wlds[128 * HID_D];   // 128 KB
    __shared__ float gatesS[64 * 128];             // 32 KB

    const int wg = blockIdx.x;
    const int tid = threadIdx.x;
    const int w = tid >> 5, lane = tid & 31, l16 = lane & 15, hi = lane >> 4;

    // --- TDM: async-load the 4 gate blocks of W_hh into LDS (2D tiles 32x512 bf16) ---
    if (w == 0) {
        unsigned ldsbase = (unsigned)(uintptr_t)(&Wlds[0]);   // low 32 bits of generic ptr = LDS offset
        for (int g = 0; g < 4; g++) {
            unsigned long long ga =
                (unsigned long long)(uintptr_t)(whh + ((size_t)g * 512 + (size_t)wg * 32) * HID_D);
            u32x4 g0;
            g0[0] = 1u;                                   // count=1, user descriptor
            g0[1] = ldsbase + (unsigned)(g * 32 * HID_D * 2);
            g0[2] = (unsigned)(ga & 0xFFFFFFFFull);
            g0[3] = (unsigned)((ga >> 32) & 0x1FFFFFFull) | (2u << 30);  // type=2 (image)
            i32x8 g1;
            g1[0] = 1 << 16;          // data_size = 2 bytes, no multicast
            g1[1] = HID_D << 16;      // tensor_dim0[15:0]
            g1[2] = 32 << 16;         // tensor_dim1[15:0] = 32 rows
            g1[3] = HID_D << 16;      // tile_dim0 = 512
            g1[4] = 32;               // tile_dim1 = 32
            g1[5] = HID_D;            // tensor_dim0_stride = 512 elements
            g1[6] = 0; g1[7] = 0;
            i32x4 z4 = {0, 0, 0, 0};
#if __clang_major__ >= 23
            i32x8 z8 = {0, 0, 0, 0, 0, 0, 0, 0};
            __builtin_amdgcn_tensor_load_to_lds(g0, g1, z4, z4, z8, 0);
#else
            __builtin_amdgcn_tensor_load_to_lds(g0, g1, z4, z4, 0);
#endif
        }
        __builtin_amdgcn_s_wait_tensorcnt(0);
    }
    __syncthreads();

    // per-thread cell state: 8 (batch, hidden) pairs
    float cst[8];
#pragma unroll
    for (int j = 0; j < 8; j++) cst[j] = 0.0f;

    const int mtile = w & 3;        // batch 16-row tile
    const int nb = (w >> 2) * 64;   // local gate-column base (0 or 64)

    for (int t = 0; t < SEQ_D; t++) {
        const unsigned short* hr = hbf + (size_t)(t & 1) * BATCH_D * HID_D;
        unsigned short* hw = hbf + (size_t)((t + 1) & 1) * BATCH_D * HID_D;

        // gates[64,128-slice] = h · W_slice^T  (K = 512, 16 iters of 32)
        v8f acc[4] = {};
        for (int kk = 0; kk < 16; kk++) {
            int kb = kk * 32 + hi * 8;
            v16bf a = ld_frag_g(hr + (size_t)(mtile * 16 + l16) * HID_D + kb);
#pragma unroll
            for (int nt = 0; nt < 4; nt++) {
                const unsigned short* bp = &Wlds[(size_t)(nb + nt * 16 + l16) * HID_D + kb];
                U256 bt;
                bt.lo = *(const u32x4*)(bp);
                bt.hi = *(const u32x4*)(bp + 16);
                acc[nt] = wmma_bf16(a, __builtin_bit_cast(v16bf, bt), acc[nt]);
            }
        }

        // stage gates (+ precomputed input projection) in LDS
#pragma unroll
        for (int nt = 0; nt < 4; nt++) {
            int nl = nb + nt * 16 + l16;
            int gcol = (nl >> 5) * 512 + wg * 32 + (nl & 31);
#pragma unroll
            for (int r = 0; r < 8; r++) {
                int b = mtile * 16 + r + hi * 8;
                gatesS[b * 128 + nl] = acc[nt][r] + xproj[((size_t)b * SEQ_D + t) * G4_D + gcol];
            }
        }
        __syncthreads();

        // elementwise cell update for our 64x32 slice (8 per thread)
#pragma unroll
        for (int j = 0; j < 8; j++) {
            int p = tid * 8 + j;
            int b = p >> 5, nn = p & 31;
            float gi = sigm_f(gatesS[b * 128 + nn]);
            float gf = sigm_f(gatesS[b * 128 + 32 + nn]);
            float gg = tanh_f(gatesS[b * 128 + 64 + nn]);
            float go = sigm_f(gatesS[b * 128 + 96 + nn]);
            float c = gf * cst[j] + gi * gg;
            cst[j] = c;
            float hv = go * tanh_f(c);
            hseq[((size_t)b * SEQ_D + t) * HID_D + wg * 32 + nn] = hv;
            hw[(size_t)b * HID_D + wg * 32 + nn] = f2bf(hv);
        }
        __threadfence();     // make h stores visible device-wide before signaling
        __syncthreads();

        // grid barrier: monotonic counter, 16 arrivals per step
        if (tid == 0) {
            __hip_atomic_fetch_add(counter, 1u, __ATOMIC_RELEASE, __HIP_MEMORY_SCOPE_AGENT);
            unsigned target = (unsigned)(NWG_LSTM * (t + 1));
            while (__hip_atomic_load(counter, __ATOMIC_ACQUIRE, __HIP_MEMORY_SCOPE_AGENT) < target) {
                __builtin_amdgcn_s_sleep(2);
            }
        }
        __syncthreads();
        __threadfence();     // acquire fresh h before next step's reads
    }
}

// ---------- FC (18 outputs) + log_softmax ----------
// block handles 32 (b,t) rows; h rows staged in LDS.
__global__ void __launch_bounds__(256) k_fc(const float* __restrict__ hseq,
                                            const float* __restrict__ wfc,
                                            const float* __restrict__ bfc,
                                            float* __restrict__ out) {
    __shared__ float hrow[32 * HID_D];          // 64 KB
    __shared__ float lg[32 * OUT_D];
    int rbase = blockIdx.x * 32;

    const f32x4* src = (const f32x4*)(hseq + (size_t)rbase * HID_D);
    for (int i = threadIdx.x; i < 32 * HID_D / 4; i += 256)
        ((f32x4*)hrow)[i] = src[i];
    __syncthreads();

    for (int p = threadIdx.x; p < 32 * OUT_D; p += 256) {
        int r = p / OUT_D, o = p % OUT_D;
        const float* wr = wfc + (size_t)o * HID_D;
        const float* hr = hrow + r * HID_D;
        float s = bfc[o];
        for (int k = 0; k < HID_D; k++) s = fmaf(hr[k], wr[k], s);
        lg[p] = s;
    }
    __syncthreads();

    if (threadIdx.x < 32) {
        int r = threadIdx.x;
        float mx = -3.0e38f;
        for (int o = 0; o < OUT_D; o++) mx = fmaxf(mx, lg[r * OUT_D + o]);
        float se = 0.0f;
        for (int o = 0; o < OUT_D; o++) se += __expf(lg[r * OUT_D + o] - mx);
        float ls = __logf(se) + mx;
        for (int o = 0; o < OUT_D; o++)
            out[(size_t)(rbase + r) * OUT_D + o] = lg[r * OUT_D + o] - ls;
    }
}

// ---------- launch ----------
extern "C" void kernel_launch(void* const* d_in, const int* in_sizes, int n_in,
                              void* d_out, int out_size, void* d_ws, size_t ws_size,
                              hipStream_t stream) {
    (void)in_sizes; (void)n_in; (void)out_size; (void)ws_size;
    const int*   x    = (const int*)d_in[0];
    const float* emb  = (const float*)d_in[1];
    const float* W_ih = (const float*)d_in[2];
    const float* W_hh = (const float*)d_in[3];
    const float* b_ih = (const float*)d_in[4];
    const float* b_hh = (const float*)d_in[5];
    const float* W_fc = (const float*)d_in[6];
    const float* b_fc = (const float*)d_in[7];
    float* out = (float*)d_out;

    char* p = (char*)d_ws;
    unsigned short* ebf  = (unsigned short*)p; p += (size_t)MTOT * EMB_D * 2;   // 16 MB
    unsigned short* wihb = (unsigned short*)p; p += (size_t)G4_D * EMB_D * 2;   // 1 MB
    unsigned short* whhb = (unsigned short*)p; p += (size_t)G4_D * HID_D * 2;   // 2 MB
    float* xproj = (float*)p;          p += (size_t)MTOT * G4_D * 4;            // 256 MB
    float* hseq  = (float*)p;          p += (size_t)MTOT * HID_D * 4;           // 64 MB
    unsigned short* hbf = (unsigned short*)p; p += (size_t)2 * BATCH_D * HID_D * 2; // 128 KB
    unsigned int* cnt = (unsigned int*)p;

    hipMemsetAsync(hbf, 0, (size_t)2 * BATCH_D * HID_D * 2, stream);
    hipMemsetAsync(cnt, 0, 256, stream);

    k_cvt_bf16<<<dim3((G4_D * EMB_D) / 256), 256, 0, stream>>>(W_ih, wihb, G4_D * EMB_D);
    k_cvt_bf16<<<dim3((G4_D * HID_D) / 256), 256, 0, stream>>>(W_hh, whhb, G4_D * HID_D);
    k_gather<<<dim3((MTOT * (EMB_D / 8)) / 256), 256, 0, stream>>>(x, emb, ebf);
    k_inproj<<<dim3(G4_D / 128, MTOT / 128), 256, 0, stream>>>(ebf, wihb, b_ih, b_hh, xproj);
    k_lstm<<<dim3(NWG_LSTM), 256, 0, stream>>>(whhb, xproj, hbf, hseq, cnt);
    k_fc<<<dim3(MTOT / 32), 256, 0, stream>>>(hseq, W_fc, b_fc, out);
}
